// CFConv_26250840113592
// MI455X (gfx1250) — matrix operand order repaired and checked by reading, not compiled, not run on hardware
//
#include <hip/hip_runtime.h>
#include <hip/hip_bf16.h>

// ---------------------------------------------------------------------------
// CDNA5 (gfx1250) fused SchNet CFConv.
// Heavy GEMMs (f_ij@Wf1, H@Wf2) run on v_wmma_f32_16x16x32_f16 (f32 accum).
// f_ij (the dominant 209 MB HBM stream) is staged into LDS by the Tensor Data
// Mover: both 128x50 f32 chunk descriptors are issued up front by wave 0, so
// chunk 1's DMA overlaps chunk 0's WMMA work (TENSORcnt in-order => wait <=1
// releases chunk 0). Weights are pre-swizzled into the wave32 WMMA B-fragment
// layout so each lane loads its 16 halves with one coalesced 32B global load.
// ---------------------------------------------------------------------------

typedef __attribute__((ext_vector_type(16))) _Float16 v16h;
typedef __attribute__((ext_vector_type(8)))  _Float16 v8h;
typedef __attribute__((ext_vector_type(8)))  float    v8f;
typedef __attribute__((ext_vector_type(4)))  unsigned int u32x4;
typedef __attribute__((ext_vector_type(8)))  int          i32x8;
typedef __attribute__((ext_vector_type(4)))  int          i32x4;

union AFrag { v16h v; v8h h8[2]; _Float16 h[16]; };

__device__ __forceinline__ float ssp(float v) {
    // shifted softplus: log(1+exp(v)) - log(2), numerically stable
    return fmaxf(v, 0.f) + __logf(1.f + __expf(-fabsf(v))) - 0.69314718056f;
}

// LDS flat addresses carry the byte offset within the workgroup's LDS in
// addr[31:0] (aperture bits live in [63:32]); truncation yields the D# lds_addr.
__device__ __forceinline__ unsigned lds_addr_of(const void* p) {
    return (unsigned)(unsigned long long)p;
}

// ---------------------------------------------------------------------------
// Issue one TDM 2D tile load: global f32 [rows_avail x row_len] (dense,
// stride row_len) -> LDS (dense), zero-filling rows >= rows_avail.
// D# built per CDNA5 ISA; groups 2/3 zero (2D tile).
// This toolchain exposes the 6-arg tensor_load_to_lds builtin
// (u32x4, i32x8, i32x4, i32x4, i32x8, i32 cpol).
// ---------------------------------------------------------------------------
__device__ __forceinline__ void tdm_load_2d_f32(const float* gptr,
                                                unsigned lds_off,
                                                unsigned rows_avail,
                                                unsigned row_len,
                                                unsigned tile_rows) {
    const unsigned long long ga = (unsigned long long)gptr;
    u32x4 g0;
    g0[0] = 1u;                                       // count=1, user mode
    g0[1] = lds_off;                                  // lds_addr (bytes)
    g0[2] = (unsigned)(ga & 0xFFFFFFFFu);             // global_addr[31:0]
    g0[3] = (unsigned)((ga >> 32) & 0x01FFFFFFu)      // global_addr[56:32]
            | (2u << 30);                             // type = 2 ("image")
    i32x8 g1;
    g1[0] = (int)(2u << 16);                          // data_size = 4 bytes
    g1[1] = (int)((row_len & 0xFFFFu) << 16);         // tensor_dim0[15:0]
    g1[2] = (int)(((row_len >> 16) & 0xFFFFu)         // tensor_dim0[31:16]
            | ((rows_avail & 0xFFFFu) << 16));        // tensor_dim1[15:0]
    g1[3] = (int)(((rows_avail >> 16) & 0xFFFFu)      // tensor_dim1[31:16]
            | ((row_len & 0xFFFFu) << 16));           // tile_dim0
    g1[4] = (int)(tile_rows & 0xFFFFu);               // tile_dim1 (tile_dim2=0)
    g1[5] = (int)row_len;                             // tensor_dim0_stride[31:0]
    g1[6] = 0;                                        // stride hi / dim1_stride
    g1[7] = 0;
    const i32x4 z4 = {0, 0, 0, 0};
    const i32x8 z8 = {0, 0, 0, 0, 0, 0, 0, 0};
    __builtin_amdgcn_tensor_load_to_lds(g0, g1, z4, z4, z8, 0);
}

// ---------------------------------------------------------------------------
// Pre-swizzle a [Ksrc x 128] f32 weight into WMMA B-fragment order (f16).
// Fragment (nt, ks): lane l, element i  <-  W[ks*32 + (i>>3)*16 + (l>>4)*8 + (i&7)][nt*16 + (l&15)]
// stored at out[((nt*ksteps + ks)*32 + l)*16 + i]. Rows >= Ksrc zero-padded.
// ---------------------------------------------------------------------------
__global__ void swizzle_w(const float* __restrict__ W, _Float16* __restrict__ out,
                          int Ksrc, int Kpad) {
    const int ksteps = Kpad >> 5;
    const int total  = 8 * ksteps * 512;
    for (int idx = blockIdx.x * blockDim.x + threadIdx.x; idx < total;
         idx += gridDim.x * blockDim.x) {
        const int i    = idx & 15;
        const int lane = (idx >> 4) & 31;
        const int frag = idx >> 9;
        const int ks   = frag % ksteps;
        const int nt   = frag / ksteps;
        const int row  = ks * 32 + ((i >> 3) << 4) + ((lane >> 4) << 3) + (i & 7);
        const int col  = nt * 16 + (lane & 15);
        out[idx] = (_Float16)((row < Ksrc) ? W[row * 128 + col] : 0.f);
    }
}

// ---------------------------------------------------------------------------
// Y[rows x 128] = X[rows x 128] @ Bsw(128x128, pre-swizzled), optional bias+ssp.
// One block = 128 rows, 8 waves x 16 rows each; K=128 -> 4 WMMA per 16x16 tile.
// ---------------------------------------------------------------------------
template <bool ACT>
__global__ __launch_bounds__(256) void lin128(const float* __restrict__ X,
                                              const _Float16* __restrict__ Bsw,
                                              const float* __restrict__ bias,
                                              float* __restrict__ Y) {
    __shared__ _Float16 sX[128][136];   // padded stride vs. bank conflicts
    const int tid  = threadIdx.x;
    const int lane = tid & 31, wave = tid >> 5;
    const long rows0 = (long)blockIdx.x * 128;

    for (int idx = tid; idx < 128 * 128; idx += 256) {
        const int r = idx >> 7, c = idx & 127;
        sX[r][c] = (_Float16)X[(rows0 + r) * 128 + c];
    }
    __syncthreads();

    const int rbase = wave * 16;
    const int col16 = lane & 15;
    const int khalf = (lane >> 4) * 8;
    const int arow  = rbase + col16;

#pragma unroll 2
    for (int nt = 0; nt < 8; ++nt) {
        v8f d = {};
#pragma unroll
        for (int ks = 0; ks < 4; ++ks) {
            AFrag a, b;
            a.h8[0] = *(const v8h*)&sX[arow][ks * 32 + khalf];
            a.h8[1] = *(const v8h*)&sX[arow][ks * 32 + 16 + khalf];
            b.v = *(const v16h*)(Bsw + ((nt * 4 + ks) * 32 + lane) * 16);
            d = __builtin_amdgcn_wmma_f32_16x16x32_f16(false, a.v, false, b.v,
                                                       (short)0, d, false, false);
        }
        const int col = nt * 16 + col16;
        const int r0  = rbase + khalf;
        const float bb = ACT ? bias[col] : 0.f;
#pragma unroll
        for (int i = 0; i < 8; ++i) {
            float v = d[i] + bb;
            if (ACT) v = ssp(v);
            Y[(rows0 + r0 + i) * 128 + col] = v;
        }
    }
}

// ---------------------------------------------------------------------------
// Main fused kernel: one block per (b, n) pair; NBH=255 in two 128-row chunks.
// TDM double-buffers the f32 f_ij chunks in LDS. Per chunk, each wave owns 16
// rows: GEMM1 (K=64, A built by f32->f16 cvt straight from the TDM tile) ->
// ssp -> f16 H in LDS (wave-local) -> GEMM2 (K=128) fused with the gather of
// y[b, nb[k], :], mask multiply and neighbor-sum into sAcc (ds_add_f32).
// ---------------------------------------------------------------------------
__global__ __launch_bounds__(256) void cfconv_main(
    const float* __restrict__ fij, const int* __restrict__ nbrs,
    const float* __restrict__ mask, const float* __restrict__ yws,
    const _Float16* __restrict__ Wf1G, const float* __restrict__ bf1,
    const _Float16* __restrict__ Wf2G, const float* __restrict__ bf2,
    float* __restrict__ agg) {
    __shared__ float    sF[2][128 * 50];   // TDM dest: f_ij chunks (dense f32)
    __shared__ _Float16 sH[128][136];      // ssp(GEMM1) result (f16)
    __shared__ int   sNb[256];
    __shared__ float sMask[256];
    __shared__ float sAcc[128];

    const int bn   = blockIdx.x;           // b*256 + n
    const int bB   = bn >> 8;
    const int tid  = threadIdx.x;
    const int lane = tid & 31, wave = tid >> 5;
    const int rbase = wave * 16;
    const int col16 = lane & 15;
    const int khalf = (lane >> 4) * 8;
    const int arow  = rbase + col16;
    const long pairBase = (long)bn * 255;

    if (tid < 128) sAcc[tid] = 0.f;
    {   // neighbor index / mask for all 255 rows (row 255 padded to zero)
        const int k = tid;
        sNb[k]   = (k < 255) ? nbrs[pairBase + k] : 0;
        sMask[k] = (k < 255) ? mask[pairBase + k] : 0.f;
    }
    if (wave == 0) {
        // Queue both chunk DMAs; TDM ops are in-order per wave, so
        // s_wait_tensorcnt<=1 releases chunk 0 while chunk 1 streams.
        tdm_load_2d_f32(fij + pairBase * 50,         lds_addr_of(&sF[0][0]),
                        128u, 50u, 128u);
        tdm_load_2d_f32(fij + (pairBase + 128) * 50, lds_addr_of(&sF[1][0]),
                        127u, 50u, 128u);   // row 127 (k=255) zero-filled
    }

    float accv[8];
#pragma unroll
    for (int i = 0; i < 8; ++i) accv[i] = 0.f;

    for (int c = 0; c < 2; ++c) {
        if (wave == 0) {
            if (c == 0) __builtin_amdgcn_s_wait_tensorcnt(1);
            else        __builtin_amdgcn_s_wait_tensorcnt(0);
        }
        __syncthreads();   // publishes TDM tile (and sNb/sMask/sAcc on c==0)

        // ---- build the two GEMM1 A-fragments straight from the f32 tile
        //      (K padded 50 -> 64; only K in [48,63] needs the k<50 guard)
        const float* rowp = &sF[c][arow * 50];
        AFrag a0, a1;
#pragma unroll
        for (int i = 0; i < 8; ++i) {
            a0.h[i]     = (_Float16)rowp[khalf + i];
            a0.h[8 + i] = (_Float16)rowp[16 + khalf + i];
            a1.h[i]     = (_Float16)rowp[32 + khalf + i];
            const int k = 48 + khalf + i;
            a1.h[8 + i] = (_Float16)((k < 50) ? rowp[k] : 0.f);
        }

        // ---- GEMM1: 16 wave-local rows, K = 64 (2 WMMA per tile)
#pragma unroll 2
        for (int nt = 0; nt < 8; ++nt) {
            v8f d = {};
            AFrag b0, b1;
            b0.v = *(const v16h*)(Wf1G + ((nt * 2 + 0) * 32 + lane) * 16);
            b1.v = *(const v16h*)(Wf1G + ((nt * 2 + 1) * 32 + lane) * 16);
            d = __builtin_amdgcn_wmma_f32_16x16x32_f16(false, a0.v, false, b0.v,
                                                       (short)0, d, false, false);
            d = __builtin_amdgcn_wmma_f32_16x16x32_f16(false, a1.v, false, b1.v,
                                                       (short)0, d, false, false);
            const int col = nt * 16 + col16;
            const float bb = bf1[col];
            const int r0 = rbase + khalf;
#pragma unroll
            for (int i = 0; i < 8; ++i)
                sH[r0 + i][col] = (_Float16)ssp(d[i] + bb);
        }
        // H rows are produced and consumed by the same wave: a dscnt drain is
        // enough (whole-wave LDS ops are in order; no cross-wave sharing).
        asm volatile("s_wait_dscnt 0" ::: "memory");

        // ---- GEMM2 (K = 128) fused with gather * mask * reduce over rows
#pragma unroll 2
        for (int nt = 0; nt < 8; ++nt) {
            v8f d = {};
#pragma unroll
            for (int ks = 0; ks < 4; ++ks) {
                AFrag a, b;
                a.h8[0] = *(const v8h*)&sH[arow][ks * 32 + khalf];
                a.h8[1] = *(const v8h*)&sH[arow][ks * 32 + 16 + khalf];
                b.v = *(const v16h*)(Wf2G + ((nt * 4 + ks) * 32 + lane) * 16);
                d = __builtin_amdgcn_wmma_f32_16x16x32_f16(false, a.v, false, b.v,
                                                           (short)0, d, false, false);
            }
            const int col = nt * 16 + col16;
            const float bb = bf2[col];
            const int r0 = c * 128 + rbase + khalf;
            float s = 0.f;
#pragma unroll
            for (int i = 0; i < 8; ++i) {
                const int r = r0 + i;
                const float yv = yws[((long)(bB * 256 + sNb[r])) * 128 + col];
                s += (d[i] + bb) * yv * sMask[r];
            }
            accv[nt] += s;
        }
    }

#pragma unroll
    for (int nt = 0; nt < 8; ++nt)
        atomicAdd(&sAcc[nt * 16 + col16], accv[nt]);   // ds_add_f32
    __syncthreads();
    if (tid < 128) agg[(long)bn * 128 + tid] = sAcc[tid];
}

// ---------------------------------------------------------------------------
// Host-side launcher. Inputs (setup_inputs order):
// 0:x 1:r_ij(unused) 2:neighbors 3:pairwise_mask 4:f_ij
// 5:Wf1 6:bf1 7:Wf2 8:bf2 9:Win 10:Wout 11:bout
// ---------------------------------------------------------------------------
extern "C" void kernel_launch(void* const* d_in, const int* in_sizes, int n_in,
                              void* d_out, int out_size, void* d_ws, size_t ws_size,
                              hipStream_t stream) {
    (void)in_sizes; (void)n_in; (void)out_size; (void)ws_size;
    const float* x    = (const float*)d_in[0];
    const int*   nbrs = (const int*)d_in[2];
    const float* mask = (const float*)d_in[3];
    const float* fij  = (const float*)d_in[4];
    const float* Wf1  = (const float*)d_in[5];
    const float* bf1  = (const float*)d_in[6];
    const float* Wf2  = (const float*)d_in[7];
    const float* bf2  = (const float*)d_in[8];
    const float* Win  = (const float*)d_in[9];
    const float* Wout = (const float*)d_in[10];
    const float* bout = (const float*)d_in[11];
    float* out = (float*)d_out;

    // workspace carve-up (~4.11 MB)
    float* y_ws = (float*)d_ws;                 // 4096*128 f32
    float* agg  = y_ws + 4096 * 128;            // 4096*128 f32
    _Float16* WinG  = (_Float16*)(agg + 4096 * 128);  // 128x128 f16, swizzled
    _Float16* WoutG = WinG  + 16384;
    _Float16* Wf1G  = WoutG + 16384;            // 64x128 f16 (padded from 50)
    _Float16* Wf2G  = Wf1G  + 8192;

    swizzle_w<<<64, 256, 0, stream>>>(Win,  WinG,  128, 128);
    swizzle_w<<<64, 256, 0, stream>>>(Wout, WoutG, 128, 128);
    swizzle_w<<<32, 256, 0, stream>>>(Wf1,  Wf1G,   50,  64);
    swizzle_w<<<64, 256, 0, stream>>>(Wf2,  Wf2G,  128, 128);

    // y = x @ Win   (4096 x 128)
    lin128<false><<<32, 256, 0, stream>>>(x, WinG, nullptr, y_ws);

    // fused filter-net + TDM-staged gather/mask/sum
    cfconv_main<<<4096, 256, 0, stream>>>(fij, nbrs, mask, y_ws,
                                          Wf1G, bf1, Wf2G, bf2, agg);

    // out = ssp(agg @ Wout + bout)
    lin128<true><<<32, 256, 0, stream>>>(agg, WoutG, bout, out);
}
